// GAT_8856222564700
// MI455X (gfx1250) — matrix operand order repaired and checked by reading, not compiled, not run on hardware
//
#include <hip/hip_runtime.h>

#define N_NODES 100000
#define E_EDGES 800000
#define EP (E_EDGES + N_NODES)   // edges incl. self loops = 900000
#define D 64
#define G_GROUPS 128

typedef __attribute__((ext_vector_type(16))) _Float16 v16h;
typedef __attribute__((ext_vector_type(8)))  float    v8f;

// ---------- order-preserving float<->uint encoding for atomicMax ----------
__device__ __forceinline__ unsigned enc_f(float f) {
    unsigned u = __float_as_uint(f);
    return (u & 0x80000000u) ? ~u : (u | 0x80000000u);
}
__device__ __forceinline__ float dec_f(unsigned u) {
    return __uint_as_float((u & 0x80000000u) ? (u ^ 0x80000000u) : ~u);
}

// ---------- per-layer init: zero accumulator, segmax, denom ----------
__global__ void zero_node_kernel(float* __restrict__ acc,
                                 unsigned* __restrict__ m,
                                 float* __restrict__ denom) {
    int i = blockIdx.x * blockDim.x + threadIdx.x;
    if (i < N_NODES * D) acc[i] = 0.0f;
    if (i < N_NODES) { m[i] = 0u; denom[i] = 0.0f; }
}

// ---------- WMMA GEMM: H[N,64] = X[N,64] @ W[64,64], plus alpha projections ----
// W is staged once per block into LDS as f16 in B-fragment layout (32B/lane,
// read back with ds_load_b128 pairs). One wave per 16-row tile: 8 v_wmma ops.
__global__ __launch_bounds__(256) void gemm_alpha_kernel(
    const float* __restrict__ X, const float* __restrict__ W,
    const float* __restrict__ a_s, const float* __restrict__ a_d,
    float* __restrict__ H, float* __restrict__ As, float* __restrict__ Ad)
{
    __shared__ v16h wlds[256];                 // [kc(2)][t(4)][lane(32)] = 8 KB

    const int tid  = threadIdx.x;
    const int lane = tid & 31;
    const int wave = tid >> 5;
    const int lr   = lane & 15;
    const int half = lane >> 4;                // 0 or 1

    // ---- cooperative fill of W fragments (all 256 threads) ----
    {
        const int fkc   = tid >> 7;            // 0..1
        const int ft    = (tid >> 5) & 3;      // 0..3
        const int flane = tid & 31;
        const int fhalf = flane >> 4;
        const int flr   = flane & 15;
        const float* wp = W + (size_t)(fkc * 32 + fhalf * 8) * D + ft * 16 + flr;
        v16h tmp;
#pragma unroll
        for (int e = 0; e < 16; ++e) tmp[e] = (_Float16)wp[(e < 8 ? e : e + 8) * D];
        wlds[tid] = tmp;
    }
    __syncthreads();

    const int tile   = blockIdx.x * 8 + wave;
    const int nTiles = N_NODES / 16;           // 6250, exact
    if (tile < nTiles) {                       // wave-uniform: EXEC all-ones below
        const int row0 = tile * 16;

        v8f acc[4];
#pragma unroll
        for (int t = 0; t < 4; ++t)
#pragma unroll
            for (int i = 0; i < 8; ++i) acc[t][i] = 0.0f;

#pragma unroll
        for (int kc = 0; kc < 2; ++kc) {
            // A fragment (16x32 f16): two contiguous 8-float runs -> b128 loads
            const float4* xv = (const float4*)(X + (size_t)(row0 + lr) * D +
                                               kc * 32 + half * 8);
            float4 p0 = xv[0], p1 = xv[1];     // K = base+0..7
            float4 p2 = xv[4], p3 = xv[5];     // K = base+16..23
            v16h a;
            a[0]  = (_Float16)p0.x; a[1]  = (_Float16)p0.y;
            a[2]  = (_Float16)p0.z; a[3]  = (_Float16)p0.w;
            a[4]  = (_Float16)p1.x; a[5]  = (_Float16)p1.y;
            a[6]  = (_Float16)p1.z; a[7]  = (_Float16)p1.w;
            a[8]  = (_Float16)p2.x; a[9]  = (_Float16)p2.y;
            a[10] = (_Float16)p2.z; a[11] = (_Float16)p2.w;
            a[12] = (_Float16)p3.x; a[13] = (_Float16)p3.y;
            a[14] = (_Float16)p3.z; a[15] = (_Float16)p3.w;

#pragma unroll
            for (int t = 0; t < 4; ++t) {
                v16h b = wlds[(kc * 4 + t) * 32 + lane];   // 2x ds_load_b128
                acc[t] = __builtin_amdgcn_wmma_f32_16x16x32_f16(
                    false, a, false, b, (short)0, acc[t], false, false);
            }
        }

        // store H and compute alpha_s/alpha_d with in-wave butterfly reductions
        float asc[4], adc[4];
#pragma unroll
        for (int t = 0; t < 4; ++t) {
            asc[t] = a_s[t * 16 + lr];
            adc[t] = a_d[t * 16 + lr];
        }
#pragma unroll
        for (int r = 0; r < 8; ++r) {
            const int row = row0 + half * 8 + r;  // lanes>=16 hold rows+8
            float ps = 0.0f, pd = 0.0f;
#pragma unroll
            for (int t = 0; t < 4; ++t) {
                float v = acc[t][r];
                H[(size_t)row * D + t * 16 + lr] = v;
                ps += v * asc[t];
                pd += v * adc[t];
            }
#pragma unroll
            for (int off = 8; off >= 1; off >>= 1) {
                ps += __shfl_xor(ps, off, 32);
                pd += __shfl_xor(pd, off, 32);
            }
            if (lr == 0) { As[row] = ps; Ad[row] = pd; }
        }
    }
}

// ---------- edge pass 1: leaky-relu logit + segment max (encoded atomicMax) ----
__global__ void edge_logit_kernel(const int* __restrict__ src, const int* __restrict__ dst,
                                  const float* __restrict__ As, const float* __restrict__ Ad,
                                  float* __restrict__ logit, unsigned* __restrict__ m) {
    int e = blockIdx.x * blockDim.x + threadIdx.x;
    if (e >= EP) return;
    int s = (e < E_EDGES) ? src[e] : (e - E_EDGES);
    int d = (e < E_EDGES) ? dst[e] : (e - E_EDGES);
    float l = As[s] + Ad[d];
    l = (l > 0.0f) ? l : 0.2f * l;
    logit[e] = l;
    atomicMax(m + d, enc_f(l));
}

// ---------- edge pass 2: e = exp(logit - max), segment sum of denominators ----
__global__ void edge_exp_kernel(const int* __restrict__ dst,
                                float* __restrict__ logit /* in: logit, out: e */,
                                const unsigned* __restrict__ m,
                                float* __restrict__ denom) {
    int e = blockIdx.x * blockDim.x + threadIdx.x;
    if (e >= EP) return;
    int d = (e < E_EDGES) ? dst[e] : (e - E_EDGES);
    float el = __expf(logit[e] - dec_f(m[d]));
    logit[e] = el;
    atomicAdd(denom + d, el);
}

// ---------- edge pass 3: acc[dst] += e * h[src]; 32 lanes x 2 comps per edge ----
__global__ void edge_scatter_kernel(const int* __restrict__ src, const int* __restrict__ dst,
                                    const float* __restrict__ el, const float* __restrict__ H,
                                    float* __restrict__ acc) {
    unsigned tid = blockIdx.x * blockDim.x + threadIdx.x;
    unsigned e = tid >> 5;
    unsigned c = (tid & 31u) * 2u;
    if (e >= EP) return;
    int s = (e < E_EDGES) ? src[e] : (int)(e - E_EDGES);
    int d = (e < E_EDGES) ? dst[e] : (int)(e - E_EDGES);
    float w = el[e];
    const float2 hv = *(const float2*)(H + (size_t)s * D + c);
    float* ap = acc + (size_t)d * D + c;
    atomicAdd(ap + 0, w * hv.x);
    atomicAdd(ap + 1, w * hv.y);
}

// ---------- finalize: out = relu(acc/denom + bias) ----------
__global__ void finalize_kernel(float* __restrict__ acc, const float* __restrict__ denom,
                                const float* __restrict__ b) {
    int i = blockIdx.x * blockDim.x + threadIdx.x;
    if (i >= N_NODES * D) return;
    int n = i >> 6;
    int c = i & 63;
    float v = acc[i] / denom[n] + b[c];
    acc[i] = (v > 0.0f) ? v : 0.0f;
}

// ---------- pooling ----------
__global__ void zero_pool_kernel(float* __restrict__ pool, float* __restrict__ cnt) {
    int i = blockIdx.x * blockDim.x + threadIdx.x;
    if (i < G_GROUPS * D) pool[i] = 0.0f;
    if (i < G_GROUPS) cnt[i] = 0.0f;
}

__global__ void pool_kernel(const float* __restrict__ H, const int* __restrict__ batch,
                            float* __restrict__ pool, float* __restrict__ cnt) {
    int i = blockIdx.x * blockDim.x + threadIdx.x;
    if (i >= N_NODES * D) return;
    int n = i >> 6;
    int c = i & 63;
    int g = batch[n];
    atomicAdd(pool + g * D + c, H[i]);
    if (c == 0) atomicAdd(cnt + g, 1.0f);
}

// ---------- head MLP: 64 -> relu 64 -> 10 per group ----------
__global__ __launch_bounds__(64) void fc_kernel(const float* __restrict__ pool,
                                                const float* __restrict__ cnt,
                                                const float* __restrict__ w1, const float* __restrict__ b1,
                                                const float* __restrict__ w2, const float* __restrict__ b2,
                                                float* __restrict__ out) {
    __shared__ float p[64];
    __shared__ float hbuf[64];
    int g = blockIdx.x;
    int j = threadIdx.x;
    float c = cnt[g];
    c = (c > 1.0f) ? c : 1.0f;
    p[j] = pool[g * 64 + j] / c;
    __syncthreads();
    float s = b1[j];
#pragma unroll 8
    for (int k = 0; k < 64; ++k) s += p[k] * w1[k * 64 + j];
    hbuf[j] = (s > 0.0f) ? s : 0.0f;
    __syncthreads();
    if (j < 10) {
        float o = b2[j];
#pragma unroll 8
        for (int k = 0; k < 64; ++k) o += hbuf[k] * w2[k * 10 + j];
        out[g * 10 + j] = o;
    }
}

extern "C" void kernel_launch(void* const* d_in, const int* in_sizes, int n_in,
                              void* d_out, int out_size, void* d_ws, size_t ws_size,
                              hipStream_t stream) {
    (void)in_sizes; (void)n_in; (void)out_size; (void)ws_size;
    const float* x     = (const float*)d_in[0];
    const int*   src   = (const int*)d_in[1];
    const int*   dst   = (const int*)d_in[2];
    const int*   batch = (const int*)d_in[3];
    const float* Wl[4]  = {(const float*)d_in[4],  (const float*)d_in[8],
                           (const float*)d_in[12], (const float*)d_in[16]};
    const float* bl[4]  = {(const float*)d_in[5],  (const float*)d_in[9],
                           (const float*)d_in[13], (const float*)d_in[17]};
    const float* asl[4] = {(const float*)d_in[6],  (const float*)d_in[10],
                           (const float*)d_in[14], (const float*)d_in[18]};
    const float* adl[4] = {(const float*)d_in[7],  (const float*)d_in[11],
                           (const float*)d_in[15], (const float*)d_in[19]};
    const float* fcw1 = (const float*)d_in[20];
    const float* fcb1 = (const float*)d_in[21];
    const float* fcw2 = (const float*)d_in[22];
    const float* fcb2 = (const float*)d_in[23];

    // workspace carving (~82 MB)
    char* ws = (char*)d_ws;
    const size_t NB = (size_t)N_NODES * D * sizeof(float);
    float*    B0    = (float*)(ws);
    float*    B1    = (float*)(ws + NB);
    float*    B2    = (float*)(ws + 2 * NB);
    float*    As    = (float*)(ws + 3 * NB);
    float*    Ad    = As + N_NODES;
    unsigned* M     = (unsigned*)(Ad + N_NODES);
    float*    denom = (float*)(M + N_NODES);
    float*    ebuf  = denom + N_NODES;
    float*    pool  = ebuf + EP;
    float*    cnt   = pool + G_GROUPS * D;

    const int nodeElems = N_NODES * D;
    const float* in = x;
    float* gemm = B0;
    float* accb[4] = {B1, B2, B1, B2};

    for (int l = 0; l < 4; ++l) {
        float* acc = accb[l];
        zero_node_kernel<<<(nodeElems + 255) / 256, 256, 0, stream>>>(acc, M, denom);
        gemm_alpha_kernel<<<(N_NODES / 16 + 7) / 8, 256, 0, stream>>>(
            in, Wl[l], asl[l], adl[l], gemm, As, Ad);
        edge_logit_kernel<<<(EP + 255) / 256, 256, 0, stream>>>(src, dst, As, Ad, ebuf, M);
        edge_exp_kernel<<<(EP + 255) / 256, 256, 0, stream>>>(dst, ebuf, M, denom);
        edge_scatter_kernel<<<((unsigned)EP * 32u + 255u) / 256u, 256, 0, stream>>>(
            src, dst, ebuf, gemm, acc);
        finalize_kernel<<<(nodeElems + 255) / 256, 256, 0, stream>>>(acc, denom, bl[l]);
        in = acc;
    }

    zero_pool_kernel<<<(G_GROUPS * D + 255) / 256, 256, 0, stream>>>(pool, cnt);
    pool_kernel<<<(nodeElems + 255) / 256, 256, 0, stream>>>(in, batch, pool, cnt);
    fc_kernel<<<G_GROUPS, 64, 0, stream>>>(pool, cnt, fcw1, fcb1, fcw2, fcb2, (float*)d_out);
}
